// OverlapCrossAttentionBlock_43705587204175
// MI455X (gfx1250) — matrix operand order repaired and checked by reading, not compiled, not run on hardware
//
#include <hip/hip_runtime.h>
#include <hip/hip_bf16.h>

typedef __attribute__((ext_vector_type(16))) _Float16 v16h;
typedef __attribute__((ext_vector_type(8)))  float    v8f;

#define TOKENS   131072      // B*H*W = 32*64*64
#define CDIM     128
#define NHEADS   4
#define HEADDIM  32
#define FFDIM    1024
#define WINDOWS  2048        // B * PY * PX = 32*8*8
#define QS_SCALE 0.17677669529663689f   // 32^-0.5
#define LN_EPS   1e-5f

// ---- CDNA5 async global->LDS copy (16 halves = 32 bytes = 2x b128), GV mode.
// LDS address: low 32 bits of the generic pointer (workgroup-relative offset).
__device__ __forceinline__ void async_copy16(void* lds_dst, const void* gsrc) {
    unsigned int       lo = (unsigned int)(size_t)lds_dst;
    unsigned long long ga = (unsigned long long)(size_t)gsrc;
    asm volatile("global_load_async_to_lds_b128 %0, %1, off\n\t"
                 "global_load_async_to_lds_b128 %0, %1, off offset:16"
                 :
                 : "v"(lo), "v"(ga)
                 : "memory");
}
// ASYNCcnt decrements in order -> partial waits enable double buffering
template <int N>
__device__ __forceinline__ void async_wait() {
    asm volatile("s_wait_asynccnt %0" :: "n"(N) : "memory");
}

union Half16 {
    uint4 q[2];
    _Float16 h[16];
};

// ---------------------------------------------------------------- f32 [K][N] -> f16 [N][K]
// weights stored K-major so all WMMA B-fragments are contiguous LDS reads
__global__ __launch_bounds__(256) void f32_to_f16_t_kernel(const float* __restrict__ src,
                                                           _Float16* __restrict__ dst,
                                                           int K, int N) {
    int i = blockIdx.x * 256 + threadIdx.x;
    if (i < K * N) {
        const int k = i / N, n = i % N;
        dst[(size_t)n * K + k] = (_Float16)src[i];
    }
}

// ---------------------------------------------------------------- expand rel-pos bias
// biasmat[head][q][k] = bias_table[rel_index[q][k]][head]
__global__ __launch_bounds__(256) void bias_expand_kernel(const float* __restrict__ bt,
                                                          const int* __restrict__ ri,
                                                          float* __restrict__ bm) {
    int i = blockIdx.x * 256 + threadIdx.x;
    if (i < NHEADS * 64 * 144) {
        const int head = i / (64 * 144);
        const int qn = i % (64 * 144);
        bm[i] = bt[(size_t)ri[qn] * NHEADS + head];
    }
}

// ---------------------------------------------------------------- LayerNorm -> f16
__global__ __launch_bounds__(256) void ln_f16_kernel(const float* __restrict__ x,
                                                     const float* __restrict__ g,
                                                     const float* __restrict__ be,
                                                     _Float16* __restrict__ out) {
    const int t = blockIdx.x * 8 + (threadIdx.x >> 5);
    const int l = threadIdx.x & 31;
    const float4 xv = *(const float4*)(x + (size_t)t * CDIM + l * 4);
    float s = xv.x + xv.y + xv.z + xv.w;
#pragma unroll
    for (int d = 16; d > 0; d >>= 1) s += __shfl_xor(s, d, 32);
    const float mu = s * (1.0f / CDIM);
    const float d0 = xv.x - mu, d1 = xv.y - mu, d2 = xv.z - mu, d3 = xv.w - mu;
    float ss = d0 * d0 + d1 * d1 + d2 * d2 + d3 * d3;
#pragma unroll
    for (int d = 16; d > 0; d >>= 1) ss += __shfl_xor(ss, d, 32);
    const float rs = rsqrtf(ss * (1.0f / CDIM) + LN_EPS);
    const float4 gv = *(const float4*)(g + l * 4);
    const float4 bv = *(const float4*)(be + l * 4);
    _Float16* o = out + (size_t)t * CDIM + l * 4;
    o[0] = (_Float16)(d0 * rs * gv.x + bv.x);
    o[1] = (_Float16)(d1 * rs * gv.y + bv.y);
    o[2] = (_Float16)(d2 * rs * gv.z + bv.z);
    o[3] = (_Float16)(d3 * rs * gv.w + bv.w);
}

// ---------------------------------------------------------------- generic f16 GEMM (K=128)
// A [T][128] row-major, BwT [N][128] K-major. 64x64 tile / WG, 4 waves.
// Full K-depth tiles staged once via async-to-LDS; all fragments contiguous b128 reads.
template <bool OUT16, bool ADD_RES>
__global__ __launch_bounds__(128) void gemm16_kernel(const _Float16* __restrict__ A,
                                                     const _Float16* __restrict__ BwT,
                                                     const float* __restrict__ bias,
                                                     const float* __restrict__ res,
                                                     float scale,
                                                     _Float16* __restrict__ out16,
                                                     float* __restrict__ out32,
                                                     int N) {
    __shared__ _Float16 As[64][128];
    __shared__ _Float16 Bt[64][128];
    const int tid = threadIdx.x;
    const int rowBase = blockIdx.x * 64;
    const int colBase = blockIdx.y * 64;
    const int w = tid >> 5, l = tid & 31;
    const int m0 = w * 16;
    const int nl = l & 15;
    const int hi = (l >> 4) & 1;

    for (int sg = tid; sg < 512; sg += 128) {
        const int r = sg >> 3, s = sg & 7;
        async_copy16(&As[r][s * 16], A + (size_t)(rowBase + r) * CDIM + s * 16);
        async_copy16(&Bt[r][s * 16], BwT + (size_t)(colBase + r) * CDIM + s * 16);
    }
    async_wait<0>();
    __syncthreads();

    const v8f vzero = {0, 0, 0, 0, 0, 0, 0, 0};
    v8f acc[4];
#pragma unroll
    for (int i = 0; i < 4; i++) acc[i] = vzero;

    const int rowm = m0 + nl;
    const int koff = hi * 8;
#pragma unroll
    for (int k0 = 0; k0 < CDIM; k0 += 32) {
        v16h a;
#pragma unroll
        for (int j = 0; j < 8; j++) {
            a[j]     = As[rowm][k0 + koff + j];
            a[8 + j] = As[rowm][k0 + 16 + koff + j];
        }
#pragma unroll
        for (int nt = 0; nt < 4; nt++) {
            v16h b;
#pragma unroll
            for (int j = 0; j < 16; j++) b[j] = Bt[nt * 16 + nl][k0 + hi * 8 + j + (j & 8)];
            acc[nt] = __builtin_amdgcn_wmma_f32_16x16x32_f16(false, a, false, b,
                                                             (short)0, acc[nt], false, false);
        }
    }
#pragma unroll
    for (int nt = 0; nt < 4; nt++) {
#pragma unroll
        for (int r = 0; r < 8; r++) {
            const int m = m0 + r + hi * 8;
            const int n = colBase + nt * 16 + nl;
            const size_t idx = (size_t)(rowBase + m) * N + n;
            float v = (acc[nt][r] + bias[n]) * scale;
            if (ADD_RES) v += res[idx];
            if (OUT16) out16[idx] = (_Float16)v;
            else       out32[idx] = v;
        }
    }
}

// ---------------------------------------------------------------- windowed attention
__global__ __launch_bounds__(128) void attn_kernel(const _Float16* __restrict__ q16,
                                                   const _Float16* __restrict__ kv16,
                                                   const float* __restrict__ bkv,
                                                   const float* __restrict__ biasmat,
                                                   _Float16* __restrict__ o16) {
    __shared__ _Float16 Qs[64][32];
    __shared__ _Float16 Ks[144][32];
    __shared__ _Float16 Vt[32][160];   // transposed V: [head_dim][key], padded keys 144..159
    __shared__ _Float16 Ps[64][160];

    const int wid  = blockIdx.x;
    const int head = blockIdx.y;
    const int b  = wid >> 6;
    const int py = (wid >> 3) & 7;
    const int px = wid & 7;
    const int tid = threadIdx.x;
    const int w = tid >> 5, l = tid & 31;
    const int m0 = w * 16;
    const int nl = l & 15;
    const int hi = (l >> 4) & 1;

    // Q tile (64x32, already scaled)
    {
        const int r = tid >> 1, s = tid & 1;
        const int my = r >> 3, mx = r & 7;
        const size_t t = (size_t)b * 4096 + (size_t)(py * 8 + my) * 64 + (px * 8 + mx);
        async_copy16(&Qs[r][s * 16], q16 + t * CDIM + head * HEADDIM + s * 16);
    }
    // K (row-major) and V (transposed) with overlap + padding -> bkv
    for (int rr = tid; rr < 288; rr += 128) {
        const int r = rr >> 1, s = rr & 1;
        const int i = r / 12, j = r % 12;
        const int ky = py * 8 - 2 + i, kx = px * 8 - 2 + j;
        if (ky >= 0 && ky < 64 && kx >= 0 && kx < 64) {
            const size_t t = (size_t)b * 4096 + (size_t)ky * 64 + kx;
            const _Float16* bp = kv16 + t * 256 + head * HEADDIM + s * 16;
            async_copy16(&Ks[r][s * 16], bp);
            Half16 u;
            u.q[0] = ((const uint4*)(bp + CDIM))[0];
            u.q[1] = ((const uint4*)(bp + CDIM))[1];
#pragma unroll
            for (int jj = 0; jj < 16; jj++) Vt[s * 16 + jj][r] = u.h[jj];
        } else {
#pragma unroll
            for (int jj = 0; jj < 16; jj++) {
                Ks[r][s * 16 + jj]  = (_Float16)bkv[head * HEADDIM + s * 16 + jj];
                Vt[s * 16 + jj][r]  = (_Float16)bkv[CDIM + head * HEADDIM + s * 16 + jj];
            }
        }
    }
    for (int z = tid; z < 512; z += 128)   // V keys 144..159 = 0
        Vt[z >> 4][144 + (z & 15)] = (_Float16)0.0f;
    for (int z = tid; z < 1024; z += 128)  // P cols 144..159 = 0
        Ps[z >> 4][144 + (z & 15)] = (_Float16)0.0f;
    async_wait<0>();
    __syncthreads();

    // S = Q K^T : 9 tiles of 16 keys, one K=32 WMMA each
    const v8f vzero = {0, 0, 0, 0, 0, 0, 0, 0};
    v8f sacc[9];
#pragma unroll
    for (int i = 0; i < 9; i++) sacc[i] = vzero;
    {
        v16h aq;
        const int rowm = m0 + nl;
        const int koff = hi * 8;
#pragma unroll
        for (int j = 0; j < 8; j++) {
            aq[j]     = Qs[rowm][koff + j];
            aq[8 + j] = Qs[rowm][16 + koff + j];
        }
#pragma unroll
        for (int nt = 0; nt < 9; nt++) {
            v16h bk;
            const int kt = nt * 16 + nl;
#pragma unroll
            for (int j = 0; j < 16; j++) bk[j] = Ks[kt][hi * 8 + j + (j & 8)];
            sacc[nt] = __builtin_amdgcn_wmma_f32_16x16x32_f16(false, aq, false, bk,
                                                              (short)0, sacc[nt], false, false);
        }
    }
    // + bias, row softmax (reduce across the 16 lanes holding one row)
    const float* bmh = biasmat + (size_t)head * 64 * 144;
#pragma unroll
    for (int r = 0; r < 8; r++) {
        const int m = m0 + r + hi * 8;
        float mx = -1e30f;
#pragma unroll
        for (int nt = 0; nt < 9; nt++) {
            const float sv = sacc[nt][r] + bmh[m * 144 + nt * 16 + nl];
            sacc[nt][r] = sv;
            mx = fmaxf(mx, sv);
        }
#pragma unroll
        for (int d = 1; d < 16; d <<= 1) mx = fmaxf(mx, __shfl_xor(mx, d, 32));
        float sum = 0.0f;
#pragma unroll
        for (int nt = 0; nt < 9; nt++) {
            const float e = __expf(sacc[nt][r] - mx);
            sacc[nt][r] = e;
            sum += e;
        }
#pragma unroll
        for (int d = 1; d < 16; d <<= 1) sum += __shfl_xor(sum, d, 32);
        const float inv = 1.0f / sum;
#pragma unroll
        for (int nt = 0; nt < 9; nt++)
            Ps[m][nt * 16 + nl] = (_Float16)(sacc[nt][r] * inv);
    }
    __syncthreads();

    // O = P @ V : K = 160 (padded), 5 chunks of 32, 2 output tiles
    v8f oacc[2];
    oacc[0] = vzero;
    oacc[1] = vzero;
#pragma unroll
    for (int kc = 0; kc < 5; kc++) {
        const int kb = kc * 32;
        v16h ap;
        const int rowm = m0 + nl;
        const int koff = hi * 8;
#pragma unroll
        for (int j = 0; j < 8; j++) {
            ap[j]     = Ps[rowm][kb + koff + j];
            ap[8 + j] = Ps[rowm][kb + 16 + koff + j];
        }
#pragma unroll
        for (int nt = 0; nt < 2; nt++) {
            v16h bv;
#pragma unroll
            for (int j = 0; j < 16; j++) bv[j] = Vt[nt * 16 + nl][kb + hi * 8 + j + (j & 8)];
            oacc[nt] = __builtin_amdgcn_wmma_f32_16x16x32_f16(false, ap, false, bv,
                                                              (short)0, oacc[nt], false, false);
        }
    }
#pragma unroll
    for (int nt = 0; nt < 2; nt++) {
#pragma unroll
        for (int r = 0; r < 8; r++) {
            const int m = m0 + r + hi * 8;
            const int my = m >> 3, mx2 = m & 7;
            const size_t t = (size_t)b * 4096 + (size_t)(py * 8 + my) * 64 + (px * 8 + mx2);
            o16[t * CDIM + head * HEADDIM + nt * 16 + nl] = (_Float16)oacc[nt][r];
        }
    }
}

// ---------------------------------------------------------------- fused MLP (double-buffered)
// w1T [FF][C] and w2T [C][FF] K-major; per 64-token tile loop FF in 64-chunks.
// Next chunk's weights stream via async-to-LDS while current chunk does WMMA;
// in-order ASYNCcnt retirement lets s_wait_asynccnt 16 fence only the older group.
__global__ __launch_bounds__(128) void mlp_kernel(const _Float16* __restrict__ h2,
                                                  const _Float16* __restrict__ w1T,
                                                  const float* __restrict__ b1,
                                                  const _Float16* __restrict__ w2T,
                                                  const float* __restrict__ b2,
                                                  const float* __restrict__ x2,
                                                  float* __restrict__ out) {
    __shared__ _Float16 Hs[64][128];
    __shared__ _Float16 W1t[2][64][128];   // ping-pong: [ff in chunk][k]
    __shared__ _Float16 W2t[2][128][64];   // ping-pong: [out col][k in chunk]
    __shared__ _Float16 Gs[64][64];

    const int tid = threadIdx.x;
    const int rowBase = blockIdx.x * 64;
    const int w = tid >> 5, l = tid & 31;
    const int m0 = w * 16;
    const int nl = l & 15;
    const int hi = (l >> 4) & 1;
    const int rowm = m0 + nl;
    const int koff = hi * 8;

    // stage H tile + chunk 0 weights (8 + 16 async instructions per thread)
    for (int sg = tid; sg < 512; sg += 128) {
        const int r = sg >> 3, s = sg & 7;
        async_copy16(&Hs[r][s * 16], h2 + (size_t)(rowBase + r) * CDIM + s * 16);
    }
    for (int sg = tid; sg < 512; sg += 128) {
        const int r = sg >> 3, s = sg & 7;
        async_copy16(&W1t[0][r][s * 16], w1T + (size_t)r * CDIM + s * 16);
        const int r2 = sg >> 2, s2 = sg & 3;
        async_copy16(&W2t[0][r2][s2 * 16], w2T + (size_t)r2 * FFDIM + s2 * 16);
    }

    const v8f vzero = {0, 0, 0, 0, 0, 0, 0, 0};
    v8f oacc[8];
#pragma unroll
    for (int i = 0; i < 8; i++) oacc[i] = vzero;

    for (int ffc = 0; ffc < 16; ffc++) {
        const int fb = ffc * 64;
        const int cur = ffc & 1;
        // prefetch next chunk into the alternate buffers (16 async instr / thread)
        if (ffc < 15) {
            const int fn = fb + 64;
            const int nxt = cur ^ 1;
            for (int sg = tid; sg < 512; sg += 128) {
                const int r = sg >> 3, s = sg & 7;
                async_copy16(&W1t[nxt][r][s * 16], w1T + (size_t)(fn + r) * CDIM + s * 16);
                const int r2 = sg >> 2, s2 = sg & 3;
                async_copy16(&W2t[nxt][r2][s2 * 16], w2T + (size_t)r2 * FFDIM + fn + s2 * 16);
            }
            async_wait<16>();   // current chunk landed; next chunk still in flight
        } else {
            async_wait<0>();
        }
        __syncthreads();

        v8f g[4];
#pragma unroll
        for (int i = 0; i < 4; i++) g[i] = vzero;
#pragma unroll
        for (int k0 = 0; k0 < CDIM; k0 += 32) {
            v16h a;
#pragma unroll
            for (int j = 0; j < 8; j++) {
                a[j]     = Hs[rowm][k0 + koff + j];
                a[8 + j] = Hs[rowm][k0 + 16 + koff + j];
            }
#pragma unroll
            for (int nt = 0; nt < 4; nt++) {
                v16h b;
#pragma unroll
                for (int j = 0; j < 16; j++) b[j] = W1t[cur][nt * 16 + nl][k0 + hi * 8 + j + (j & 8)];
                g[nt] = __builtin_amdgcn_wmma_f32_16x16x32_f16(false, a, false, b,
                                                               (short)0, g[nt], false, false);
            }
        }
        // exact GELU -> Gs (f16)
#pragma unroll
        for (int nt = 0; nt < 4; nt++) {
#pragma unroll
            for (int r = 0; r < 8; r++) {
                const int m = m0 + r + hi * 8;
                const int n = nt * 16 + nl;
                const float v = g[nt][r] + b1[fb + n];
                Gs[m][n] = (_Float16)(0.5f * v * (1.0f + erff(v * 0.70710678118654752f)));
            }
        }
        __syncthreads();
        // O += G @ W2chunk
#pragma unroll
        for (int k2 = 0; k2 < 64; k2 += 32) {
            v16h a;
#pragma unroll
            for (int j = 0; j < 8; j++) {
                a[j]     = Gs[rowm][k2 + koff + j];
                a[8 + j] = Gs[rowm][k2 + 16 + koff + j];
            }
#pragma unroll
            for (int nt = 0; nt < 8; nt++) {
                v16h b;
#pragma unroll
                for (int j = 0; j < 16; j++) b[j] = W2t[cur][nt * 16 + nl][k2 + hi * 8 + j + (j & 8)];
                oacc[nt] = __builtin_amdgcn_wmma_f32_16x16x32_f16(false, a, false, b,
                                                                  (short)0, oacc[nt], false, false);
            }
        }
        __syncthreads();
    }
#pragma unroll
    for (int nt = 0; nt < 8; nt++) {
#pragma unroll
        for (int r = 0; r < 8; r++) {
            const int m = m0 + r + hi * 8;
            const int n = nt * 16 + nl;
            const size_t idx = (size_t)(rowBase + m) * CDIM + n;
            out[idx] = oacc[nt][r] + b2[n] + x2[idx];
        }
    }
}

// ---------------------------------------------------------------- launcher
extern "C" void kernel_launch(void* const* d_in, const int* in_sizes, int n_in,
                              void* d_out, int out_size, void* d_ws, size_t ws_size,
                              hipStream_t stream) {
    (void)in_sizes; (void)n_in; (void)out_size; (void)ws_size;
    const float* x          = (const float*)d_in[0];
    const float* wq         = (const float*)d_in[1];
    const float* bq         = (const float*)d_in[2];
    const float* wkv        = (const float*)d_in[3];
    const float* bkv        = (const float*)d_in[4];
    const float* wproj      = (const float*)d_in[5];
    const float* bproj      = (const float*)d_in[6];
    const float* bias_table = (const float*)d_in[7];
    const float* g1         = (const float*)d_in[8];
    const float* be1        = (const float*)d_in[9];
    const float* g2         = (const float*)d_in[10];
    const float* be2        = (const float*)d_in[11];
    const float* w_mlp1     = (const float*)d_in[12];
    const float* b_mlp1     = (const float*)d_in[13];
    const float* w_mlp2     = (const float*)d_in[14];
    const float* b_mlp2     = (const float*)d_in[15];
    const int*   rel_index  = (const int*)d_in[16];
    float* out = (float*)d_out;

    char* p = (char*)d_ws;
    auto alloc = [&](size_t bytes) -> char* {
        char* r = p;
        p += (bytes + 255) & ~(size_t)255;
        return r;
    };
    const size_t T = TOKENS;
    _Float16* wqT    = (_Float16*)alloc(128 * 128 * 2);
    _Float16* wkvT   = (_Float16*)alloc(256 * 128 * 2);
    _Float16* wprojT = (_Float16*)alloc(128 * 128 * 2);
    _Float16* w1T    = (_Float16*)alloc(1024 * 128 * 2);
    _Float16* w2T    = (_Float16*)alloc(128 * 1024 * 2);
    float*    biasmat= (float*)alloc(NHEADS * 64 * 144 * 4);
    _Float16* h16    = (_Float16*)alloc(T * CDIM * 2);
    _Float16* q16    = (_Float16*)alloc(T * CDIM * 2);
    _Float16* kv16   = (_Float16*)alloc(T * 256 * 2);
    _Float16* o16    = (_Float16*)alloc(T * CDIM * 2);
    _Float16* h2_16  = (_Float16*)alloc(T * CDIM * 2);
    float*    x2     = (float*)alloc(T * CDIM * 4);

    // weights -> f16, K-major (transposed)
    f32_to_f16_t_kernel<<<(128 * 128 + 255) / 256, 256, 0, stream>>>(wq, wqT, 128, 128);
    f32_to_f16_t_kernel<<<(128 * 256 + 255) / 256, 256, 0, stream>>>(wkv, wkvT, 128, 256);
    f32_to_f16_t_kernel<<<(128 * 128 + 255) / 256, 256, 0, stream>>>(wproj, wprojT, 128, 128);
    f32_to_f16_t_kernel<<<(128 * 1024 + 255) / 256, 256, 0, stream>>>(w_mlp1, w1T, 128, 1024);
    f32_to_f16_t_kernel<<<(1024 * 128 + 255) / 256, 256, 0, stream>>>(w_mlp2, w2T, 1024, 128);
    // expanded attention bias
    bias_expand_kernel<<<(NHEADS * 64 * 144 + 255) / 256, 256, 0, stream>>>(
        bias_table, rel_index, biasmat);

    // LN1
    ln_f16_kernel<<<T / 8, 256, 0, stream>>>(x, g1, be1, h16);
    // Q = (h @ wq + bq) * SCALE
    gemm16_kernel<true, false><<<dim3(T / 64, 2), 128, 0, stream>>>(
        h16, wqT, bq, nullptr, QS_SCALE, q16, nullptr, 128);
    // KV = h @ wkv + bkv (full grid; windows gathered in attention)
    gemm16_kernel<true, false><<<dim3(T / 64, 4), 128, 0, stream>>>(
        h16, wkvT, bkv, nullptr, 1.0f, kv16, nullptr, 256);
    // attention
    attn_kernel<<<dim3(WINDOWS, NHEADS), 128, 0, stream>>>(q16, kv16, bkv, biasmat, o16);
    // x2 = o @ wproj + bproj + x
    gemm16_kernel<false, true><<<dim3(T / 64, 2), 128, 0, stream>>>(
        o16, wprojT, bproj, x, 1.0f, nullptr, x2, 128);
    // LN2
    ln_f16_kernel<<<T / 8, 256, 0, stream>>>(x2, g2, be2, h2_16);
    // fused MLP + residual
    mlp_kernel<<<T / 64, 128, 0, stream>>>(h2_16, w1T, b_mlp1, w2T, b_mlp2, x2, out);
}
